// GCNModel_75874892251953
// MI455X (gfx1250) — matrix operand order repaired and checked by reading, not compiled, not run on hardware
//
#include <hip/hip_runtime.h>
#include <hip/hip_bf16.h>

typedef __attribute__((ext_vector_type(16))) __bf16 v16bf;
typedef __attribute__((ext_vector_type(8)))  float  v8f;

#define N_NODES 50000
#define N_EDGES 800000
#define IN_CH   64
#define HID     128
#define LAST_HID 256

// ---------------- init: zero accumulators, seed z with fc1_b ----------------
__global__ void init_ker(float* __restrict__ summed, float* __restrict__ deg,
                         float* __restrict__ psum,  float* __restrict__ z,
                         const float* __restrict__ fc1_b) {
    const int NS = N_NODES * IN_CH;            // 3,200,000
    const int total = NS + N_NODES + N_NODES + LAST_HID;
    for (int idx = blockIdx.x * blockDim.x + threadIdx.x; idx < total;
         idx += gridDim.x * blockDim.x) {
        if (idx < NS)                       summed[idx] = 0.0f;
        else if (idx < NS + N_NODES)        deg[idx - NS] = 0.0f;
        else if (idx < NS + 2 * N_NODES)    psum[idx - NS - N_NODES] = 0.0f;
        else                                z[idx - NS - 2 * N_NODES] = fc1_b[idx - NS - 2 * N_NODES];
    }
}

// ---------------- layer-1 edge scatter: segment_sum(x[src]) + degree -------
__global__ void scatter1_ker(const int* __restrict__ ei, const float* __restrict__ x,
                             float* __restrict__ summed, float* __restrict__ deg) {
    int tid = blockIdx.x * blockDim.x + threadIdx.x;   // N_EDGES * 8 threads
    if (tid >= N_EDGES * 8) return;
    int e = tid >> 3, g = tid & 7;
    int src = ei[e];
    int dst = ei[N_EDGES + e];
    const float* xs = x + (size_t)src * IN_CH + g * 8;
    float* sd = summed + (size_t)dst * IN_CH + g * 8;
    #pragma unroll
    for (int c = 0; c < 8; ++c) atomicAdd(&sd[c], xs[c]);
    if (g == 0) atomicAdd(&deg[dst], 1.0f);
}

// ---------------- layer-1 dual GEMM via WMMA: h1 = relu(agg@W1l^T + b + x@W1r^T)
__global__ __launch_bounds__(32) void sage1_wmma_ker(
    const float* __restrict__ x,      // N_NODES x 64
    const float* __restrict__ summed, // N_NODES x 64
    const float* __restrict__ deg,    // N_NODES
    const float* __restrict__ W1l,    // 128 x 64
    const float* __restrict__ b1l,    // 128
    const float* __restrict__ W1r,    // 128 x 64
    float* __restrict__ h1)           // N_NODES x 128
{
    const int tile  = blockIdx.x;          // 3125 * 8 tiles
    const int mTile = tile >> 3;
    const int nTile = tile & 7;
    const int lane  = threadIdx.x;         // wave32, EXEC all-ones
    const int m     = lane & 15;
    const int hi    = lane >> 4;
    const int row   = mTile * 16 + m;                 // A: node row
    const int ncol  = nTile * 16 + (lane & 15);       // B/C: output channel
    const float invdeg = 1.0f / fmaxf(deg[row], 1.0f);

    v8f acc = {};
    #pragma unroll
    for (int kc = 0; kc < 2; ++kc) {
        v16bf aAgg, aX, bL, bR;
        #pragma unroll
        for (int j = 0; j < 16; ++j) {
            // A-matrix 16x32 bf16 layout: lane m holds K = base + hi*8 + (j&7), (j<8 -> +0, else +16)
            int kA = kc * 32 + ((j < 8) ? 0 : 16) + hi * 8 + (j & 7);
            aAgg[j] = (__bf16)(summed[row * IN_CH + kA] * invdeg);
            aX[j]   = (__bf16)(x[row * IN_CH + kA]);
            // B-matrix 32x16 bf16 layout: same K order as A: K = base + hi*8 + (j&7), (j<8 -> +0, else +16)
            int kB = kc * 32 + ((j < 8) ? 0 : 16) + hi * 8 + (j & 7);
            bL[j] = (__bf16)(W1l[ncol * IN_CH + kB]);
            bR[j] = (__bf16)(W1r[ncol * IN_CH + kB]);
        }
        acc = __builtin_amdgcn_wmma_f32_16x16x32_bf16(false, aAgg, false, bL,
                                                      (short)0, acc, false, false);
        acc = __builtin_amdgcn_wmma_f32_16x16x32_bf16(false, aX,   false, bR,
                                                      (short)0, acc, false, false);
    }
    const float bias = b1l[ncol];
    #pragma unroll
    for (int r = 0; r < 8; ++r) {
        int node = mTile * 16 + r + hi * 8;      // C/D: VGPR r -> M = r + 8*hi
        float val = acc[r] + bias;
        h1[(size_t)node * HID + ncol] = val > 0.0f ? val : 0.0f;
    }
}

// ---------------- layer-2 projections: p = h1@W2l^T, q = h1@W2r^T ----------
__global__ void sage2_proj_ker(const float* __restrict__ h1,
                               const float* __restrict__ W2l,
                               const float* __restrict__ W2r,
                               float* __restrict__ pnode, float* __restrict__ qnode) {
    __shared__ float wl[HID], wr[HID];
    for (int i = threadIdx.x; i < HID; i += blockDim.x) { wl[i] = W2l[i]; wr[i] = W2r[i]; }
    __syncthreads();
    int n = blockIdx.x * blockDim.x + threadIdx.x;
    if (n >= N_NODES) return;
    const float* hr = h1 + (size_t)n * HID;
    float p = 0.0f, q = 0.0f;
    #pragma unroll 8
    for (int c = 0; c < HID; ++c) { float hv = hr[c]; p += hv * wl[c]; q += hv * wr[c]; }
    pnode[n] = p; qnode[n] = q;
}

// ---------------- layer-2 scalar edge scatter ------------------------------
__global__ void scatter2_ker(const int* __restrict__ ei, const float* __restrict__ pnode,
                             float* __restrict__ psum) {
    int e = blockIdx.x * blockDim.x + threadIdx.x;
    if (e >= N_EDGES) return;
    atomicAdd(&psum[ei[N_EDGES + e]], pnode[ei[e]]);
}

// ---------------- v[n] = relu(psum/deg + b2l + q) --------------------------
__global__ void vker(const float* __restrict__ psum, const float* __restrict__ qnode,
                     const float* __restrict__ deg, const float* __restrict__ b2l,
                     float* __restrict__ v) {
    int n = blockIdx.x * blockDim.x + threadIdx.x;
    if (n >= N_NODES) return;
    float invdeg = 1.0f / fmaxf(deg[n], 1.0f);
    float val = psum[n] * invdeg + b2l[0] + qnode[n];
    v[n] = val > 0.0f ? val : 0.0f;
}

// ---------------- z = fc1_W @ v (+fc1_b preloaded), split-K ----------------
__global__ void fc1_ker(const float* __restrict__ W, const float* __restrict__ v,
                        float* __restrict__ z) {
    __shared__ float red[256];
    const int j = blockIdx.x;                       // output row 0..255
    const int chunk = (N_NODES + gridDim.y - 1) / gridDim.y;
    const int i0 = blockIdx.y * chunk;
    int i1 = i0 + chunk; if (i1 > N_NODES) i1 = N_NODES;
    float s = 0.0f;
    for (int i = i0 + threadIdx.x; i < i1; i += blockDim.x)
        s += W[(size_t)j * N_NODES + i] * v[i];
    red[threadIdx.x] = s;
    __syncthreads();
    for (int off = 128; off > 0; off >>= 1) {
        if (threadIdx.x < off) red[threadIdx.x] += red[threadIdx.x + off];
        __syncthreads();
    }
    if (threadIdx.x == 0) atomicAdd(&z[j], red[0]);
}

// ---------------- pred = fc2_W @ z + fc2_b ---------------------------------
__global__ void fc2_ker(const float* __restrict__ fc2_W, const float* __restrict__ fc2_b,
                        const float* __restrict__ z, float* __restrict__ out) {
    __shared__ float red[LAST_HID];
    red[threadIdx.x] = fc2_W[threadIdx.x] * z[threadIdx.x];
    __syncthreads();
    for (int off = LAST_HID / 2; off > 0; off >>= 1) {
        if (threadIdx.x < off) red[threadIdx.x] += red[threadIdx.x + off];
        __syncthreads();
    }
    if (threadIdx.x == 0) out[0] = red[0] + fc2_b[0];
}

extern "C" void kernel_launch(void* const* d_in, const int* in_sizes, int n_in,
                              void* d_out, int out_size, void* d_ws, size_t ws_size,
                              hipStream_t stream) {
    const float* x     = (const float*)d_in[0];
    const int*   ei    = (const int*)  d_in[1];   // [2][N_EDGES]
    const float* W1l   = (const float*)d_in[2];
    const float* b1l   = (const float*)d_in[3];
    const float* W1r   = (const float*)d_in[4];
    const float* W2l   = (const float*)d_in[5];
    const float* b2l   = (const float*)d_in[6];
    const float* W2r   = (const float*)d_in[7];
    const float* fc1_W = (const float*)d_in[8];
    const float* fc1_b = (const float*)d_in[9];
    const float* fc2_W = (const float*)d_in[10];
    const float* fc2_b = (const float*)d_in[11];

    float* ws     = (float*)d_ws;
    float* summed = ws;                          // 3,200,000
    float* deg    = ws + 3200000;                //    50,000
    float* h1     = ws + 3250000;                // 6,400,000
    float* pnode  = ws + 9650000;                //    50,000
    float* qnode  = ws + 9700000;                //    50,000
    float* psum   = ws + 9750000;                //    50,000
    float* z      = ws + 9800000;                //       256
    float* vvec   = pnode;                       // reuse after scatter2

    init_ker<<<12892, 256, 0, stream>>>(summed, deg, psum, z, fc1_b);
    scatter1_ker<<<(N_EDGES * 8 + 255) / 256, 256, 0, stream>>>(ei, x, summed, deg);
    sage1_wmma_ker<<<(N_NODES / 16) * (HID / 16), 32, 0, stream>>>(
        x, summed, deg, W1l, b1l, W1r, h1);
    sage2_proj_ker<<<(N_NODES + 255) / 256, 256, 0, stream>>>(h1, W2l, W2r, pnode, qnode);
    scatter2_ker<<<(N_EDGES + 255) / 256, 256, 0, stream>>>(ei, pnode, psum);
    vker<<<(N_NODES + 255) / 256, 256, 0, stream>>>(psum, qnode, deg, b2l, vvec);
    fc1_ker<<<dim3(256, 25), 256, 0, stream>>>(fc1_W, vvec, z);
    fc2_ker<<<1, LAST_HID, 0, stream>>>(fc2_W, fc2_b, z, (float*)d_out);
}
